// MultiHeadAttention_86517821210964
// MI455X (gfx1250) — matrix operand order repaired and checked by reading, not compiled, not run on hardware
//
#include <hip/hip_runtime.h>

typedef __attribute__((ext_vector_type(16))) _Float16 v16h;
typedef __attribute__((ext_vector_type(8)))  float    v8f;
typedef __attribute__((ext_vector_type(4)))  int      v4i_;

#define B_   4
#define L_   1024
#define DM   1024
#define NH   16
#define DK   64
#define MTOT (B_ * L_)   // 4096

// ------------- gfx1250 async global->LDS copy (guarded) -------------------
#if defined(__has_builtin)
#  if __has_builtin(__builtin_amdgcn_global_load_async_to_lds_b128) && \
      __has_builtin(__builtin_amdgcn_s_wait_asynccnt)
#    define USE_ASYNC_COPY 1
#  endif
#endif
#ifndef USE_ASYNC_COPY
#  define USE_ASYNC_COPY 0
#endif

// Copy 16 bytes global -> LDS (per-lane addresses), ASYNCcnt-tracked if possible.
__device__ __forceinline__ void copy_b128_g2l(const void* g, void* l) {
#if USE_ASYNC_COPY
  typedef __attribute__((address_space(1))) v4i_ gvec_t;   // global int4
  typedef __attribute__((address_space(3))) v4i_ lvec_t;   // LDS int4
  __builtin_amdgcn_global_load_async_to_lds_b128((gvec_t*)g, (lvec_t*)l, 0, 0);
#else
  *(v4i_*)l = *(const v4i_*)g;
#endif
}

// Wait for this wave's async copies, then workgroup barrier.
__device__ __forceinline__ void copy_wait_barrier() {
#if USE_ASYNC_COPY
  __builtin_amdgcn_s_wait_asynccnt(0);
#endif
  __syncthreads();
}

// ---------- WMMA fragment loaders (16x16x32 f16, wave32 layouts) ----------
// A (16x32, MxK): lanes hold row M=lane&15, K = kb..kb+15 contiguous
// with kb = k0 + (lane<16 ? 0 : 16).
__device__ __forceinline__ v16h frag_a_ldsf16(const _Float16* s, int ld, int k0, int lane) {
  int r  = lane & 15;
  int kb = k0 + ((lane >> 4) << 4);
  const _Float16* p = s + r * ld + kb;
  v16h a;
#pragma unroll
  for (int i = 0; i < 8; ++i) a[i]     = p[i];
#pragma unroll
  for (int i = 0; i < 8; ++i) a[i + 8] = p[i + 8];
  return a;
}

__device__ __forceinline__ v16h frag_a_ldsf32(const float* s, int ld, int k0, int lane) {
  int r  = lane & 15;
  int kb = k0 + ((lane >> 4) << 4);
  const float* p = s + r * ld + kb;
  v16h a;
#pragma unroll
  for (int i = 0; i < 8; ++i) a[i]     = (_Float16)p[i];
#pragma unroll
  for (int i = 0; i < 8; ++i) a[i + 8] = (_Float16)p[i + 8];
  return a;
}

// B (32x16, KxN) sourced from an [N,K] row-major array: lane holds column
// N = n0 + (lane&15), K = k0 + (lane<16 ? 0 : 16) .. +15, contiguous.
__device__ __forceinline__ v16h frag_b_f32g(const float* __restrict__ w, int ld,
                                            int n0, int k0, int lane) {
  int n  = lane & 15;
  int kb = k0 + ((lane >> 4) << 4);
  const float* p = w + (size_t)(n0 + n) * ld + kb;
  __builtin_prefetch(p + 32, 0, 0);          // next K-step of this weight row
  v16h b;
#pragma unroll
  for (int i = 0; i < 16; ++i) b[i] = (_Float16)p[i];
  return b;
}

__device__ __forceinline__ v16h frag_b_f16g(const _Float16* __restrict__ w, int ld,
                                            int n0, int k0, int lane) {
  int n  = lane & 15;
  int kb = k0 + ((lane >> 4) << 4);
  const _Float16* p = w + (size_t)(n0 + n) * ld + kb;
  v16h b;
#pragma unroll
  for (int i = 0; i < 16; ++i) b[i] = p[i];
  return b;
}

// ---------------- Kernel 1: fused QKV projection GEMMs -------------------
// One block = one (matrix, 16-row block); A row-block async-staged to LDS
// once, converted to f16 once, reused by all 8 waves x 8 n-tiles.
// qh/kh stored [B,H,L,64] f16; v stored transposed [B,H,64,L] f16.
__global__ __launch_bounds__(256)
void qkv_proj_kernel(const float* __restrict__ q, const float* __restrict__ k,
                     const float* __restrict__ v,
                     const float* __restrict__ wq, const float* __restrict__ bq,
                     const float* __restrict__ wk, const float* __restrict__ bk,
                     const float* __restrict__ wv, const float* __restrict__ bv,
                     _Float16* __restrict__ qh, _Float16* __restrict__ kh,
                     _Float16* __restrict__ vhT) {
  extern __shared__ char smem[];
  float*    Af = (float*)smem;                 // 16*1024 f32 = 64 KB
  _Float16* Ah = (_Float16*)(smem + 65536);    // 16*1024 f16 = 32 KB

  int t = threadIdx.x;
  int wave = t >> 5, lane = t & 31;
  int mat   = blockIdx.x >> 8;                 // 0:q 1:k 2:v (block-uniform)
  int mtile = blockIdx.x & 255;
  int m0    = mtile << 4;

  const float* X  = (mat == 0) ? q  : (mat == 1) ? k  : v;
  const float* W  = (mat == 0) ? wq : (mat == 1) ? wk : wv;
  const float* BI = (mat == 0) ? bq : (mat == 1) ? bk : bv;
  // Uniform store base + strides: qh/kh are [.., l, d] (ls=DK, ds=1),
  // vhT is [.., d, l] (ls=1, ds=L_).  Per-(b,h) block size is L_*DK either way.
  _Float16* DST = (mat == 0) ? qh : (mat == 1) ? kh : vhT;
  int ls = (mat == 2) ? 1 : DK;
  int ds = (mat == 2) ? L_ : 1;

  // --- async-stage 64 KB contiguous A row-block (rows m0..m0+15) ---
  const float* gA = X + (size_t)m0 * DM;
#pragma unroll
  for (int i = 0; i < 16; ++i) {
    int e = (t + i * 256) << 2;                // 16-byte chunk
    copy_b128_g2l(gA + e, Af + e);
  }
  copy_wait_barrier();

  // --- convert once to f16 (pair-stride: conflict-free, pk-convert) ---
#pragma unroll
  for (int i = 0; i < 32; ++i) {
    int e = (t + i * 256) << 1;
    Ah[e]     = (_Float16)Af[e];
    Ah[e + 1] = (_Float16)Af[e + 1];
  }
  __syncthreads();

  // --- 8 n-tiles per wave ---
  for (int j = 0; j < 8; ++j) {
    int n0 = ((wave * 8) + j) << 4;
    v8f acc = {};
    for (int k0 = 0; k0 < DM; k0 += 32) {
      v16h a = frag_a_ldsf16(Ah, DM, k0, lane);
      v16h b = frag_b_f32g(W, DM, n0, k0, lane);
      acc = __builtin_amdgcn_wmma_f32_16x16x32_f16(false, a, false, b,
                                                   (short)0, acc, false, false);
    }
    int nloc = lane & 15;
    int n    = n0 + nloc;
    float bias = BI[n];
    int h = n >> 6, d = n & 63;
    int mbase = m0 + ((lane >> 4) << 3);
#pragma unroll
    for (int r = 0; r < 8; ++r) {
      int m = mbase + r;
      int bb = m >> 10, l = m & 1023;
      DST[(size_t)(bb * NH + h) * (L_ * DK) + l * ls + d * ds] =
          (_Float16)(acc[r] + bias);
    }
  }
}

// -------- Kernel 2: fused scores + mask + softmax + attn@V ----------------
// One block per (h, b, 16 query rows). Q tile async-staged to LDS; score
// tile [16][1024] lives in LDS; attn (fp32) streamed to d_out exactly once.
// Elementwise phases use k = cc + i*16 so global accesses are coalesced and
// LDS conflicts are <= 2-way.
__global__ __launch_bounds__(256)
void attn_kernel(const _Float16* __restrict__ qh, const _Float16* __restrict__ kh,
                 const _Float16* __restrict__ vhT,
                 const unsigned char* __restrict__ mask,
                 float* __restrict__ attn_out, _Float16* __restrict__ ctx) {
  extern __shared__ char smem[];
  float*    sc = (float*)smem;                 // 16*1024 f32 = 64 KB
  _Float16* Qs = (_Float16*)(smem + 65536);    // 16*64 f16 = 2 KB

  int t = threadIdx.x;
  int wave = t >> 5, lane = t & 31;
  int bi = blockIdx.x;                         // 16*4*64 = 4096
  int h  = bi >> 8;
  int b  = (bi >> 6) & 3;
  int q0 = (bi & 63) << 4;

  const _Float16* Q  = qh  + ((size_t)(b * NH + h)) * L_ * DK;
  const _Float16* Kh = kh  + ((size_t)(b * NH + h)) * L_ * DK;
  const _Float16* Vt = vhT + ((size_t)(b * NH + h)) * DK * L_;

  // --- stage 2 KB contiguous Q tile (rows q0..q0+15) into LDS ---
  if (t < 128) {
    int e = t << 3;                            // 8 halves = 16 bytes
    copy_b128_g2l(Q + (size_t)q0 * DK + e, Qs + e);
  }
  copy_wait_barrier();

  // --- scores = Q K^T / 8 : each wave computes 8 of the 64 k-tiles ---
  for (int kt = wave * 8; kt < wave * 8 + 8; ++kt) {
    int n0 = kt << 4;
    v8f acc = {};
#pragma unroll
    for (int d0 = 0; d0 < DK; d0 += 32) {
      v16h a = frag_a_ldsf16(Qs, DK, d0, lane);
      v16h c = frag_b_f16g(Kh, DK, n0, d0, lane);
      acc = __builtin_amdgcn_wmma_f32_16x16x32_f16(false, a, false, c,
                                                   (short)0, acc, false, false);
    }
    int nloc  = lane & 15;
    int mbase = (lane >> 4) << 3;
#pragma unroll
    for (int r = 0; r < 8; ++r)
      sc[(mbase + r) * 1024 + n0 + nloc] = acc[r] * 0.125f;
  }
  __syncthreads();

  // --- mask + softmax: thread (row rr, lane-slot cc) owns k = cc + 16*i ---
  int rr = t >> 4;
  int cc = t & 15;
  float* row = sc + rr * 1024;
  const unsigned char* mrow = mask + ((size_t)b * L_ + (q0 + rr)) * L_;
  const float NEG_INF = -__builtin_inff();

  float mx = NEG_INF;
  for (int i = 0; i < 64; ++i) {
    int kk = cc + (i << 4);
    float s = mrow[kk] ? NEG_INF : row[kk];
    row[kk] = s;
    mx = fmaxf(mx, s);
  }
  for (int o = 8; o; o >>= 1) mx = fmaxf(mx, __shfl_xor(mx, o, 32));

  float sum = 0.f;
  for (int i = 0; i < 64; ++i) {
    int kk = cc + (i << 4);
    float e = __expf(row[kk] - mx);
    row[kk] = e;
    sum += e;
  }
  for (int o = 8; o; o >>= 1) sum += __shfl_xor(sum, o, 32);
  float inv = 1.0f / sum;

  float* orow = attn_out + (((size_t)(h * B_ + b)) * L_ + (q0 + rr)) * L_;
  for (int i = 0; i < 64; ++i) {
    int kk = cc + (i << 4);
    float p = row[kk] * inv;
    row[kk] = p;
    orow[kk] = p;                              // coalesced fp32 attn store
  }
  __syncthreads();

  // --- ctx = attn @ V : waves 0-3 each own one 16-wide slice of Dv=64 ---
  if (wave < 4) {
    int n0 = wave << 4;
    v8f acc = {};
    for (int k0 = 0; k0 < L_; k0 += 32) {
      v16h a = frag_a_ldsf32(sc, 1024, k0, lane);
      v16h c = frag_b_f16g(Vt, L_, n0, k0, lane);
      acc = __builtin_amdgcn_wmma_f32_16x16x32_f16(false, a, false, c,
                                                   (short)0, acc, false, false);
    }
    int nloc  = lane & 15;
    int mbase = (lane >> 4) << 3;
#pragma unroll
    for (int r = 0; r < 8; ++r) {
      int m = q0 + mbase + r;
      ctx[((size_t)(b * L_ + m)) * DM + h * DK + n0 + nloc] = (_Float16)acc[r];
    }
  }
}

// ------ Kernel 3: FC projection + bias + residual + LayerNorm ------------
__global__ __launch_bounds__(256)
void fc_ln_kernel(const _Float16* __restrict__ ctx, const float* __restrict__ fcw,
                  const float* __restrict__ fcb, const float* __restrict__ resid,
                  const float* __restrict__ lng, const float* __restrict__ lnb,
                  float* __restrict__ y) {
  extern __shared__ char smem[];
  float*    xs = (float*)smem;                 // 16*1024 f32 = 64 KB
  _Float16* Ah = (_Float16*)(smem + 65536);    // 16*1024 f16 = 32 KB

  int t = threadIdx.x;
  int wave = t >> 5, lane = t & 31;
  int m0 = blockIdx.x << 4;

  // --- async-stage 32 KB contiguous ctx row-block (already f16) ---
  const _Float16* gA = ctx + (size_t)m0 * DM;
#pragma unroll
  for (int i = 0; i < 8; ++i) {
    int e = (t + i * 256) << 3;                // 8 halves = 16 bytes
    copy_b128_g2l(gA + e, Ah + e);
  }
  copy_wait_barrier();

  for (int j = 0; j < 8; ++j) {
    int n0 = ((wave * 8) + j) << 4;
    v8f acc = {};
    for (int k0 = 0; k0 < DM; k0 += 32) {
      v16h a = frag_a_ldsf16(Ah, DM, k0, lane);
      v16h b = frag_b_f32g(fcw, DM, n0, k0, lane);
      acc = __builtin_amdgcn_wmma_f32_16x16x32_f16(false, a, false, b,
                                                   (short)0, acc, false, false);
    }
    int nloc  = lane & 15;
    int n     = n0 + nloc;
    float bias = fcb[n];
    int mbase = (lane >> 4) << 3;
#pragma unroll
    for (int r = 0; r < 8; ++r) {
      int m = m0 + mbase + r;
      xs[(mbase + r) * 1024 + n] = acc[r] + bias + resid[(size_t)m * DM + n];
    }
  }
  __syncthreads();

  // --- LayerNorm: interleaved k = cc + 16*i (coalesced, low-conflict) ---
  int rr = t >> 4;
  int cc = t & 15;
  float* row = xs + rr * 1024;
  float s = 0.f, sq = 0.f;
  for (int i = 0; i < 64; ++i) {
    float xv = row[cc + (i << 4)];
    s += xv; sq += xv * xv;
  }
  for (int o = 8; o; o >>= 1) { s  += __shfl_xor(s,  o, 32);
                                sq += __shfl_xor(sq, o, 32); }
  float mu   = s * (1.0f / 1024.0f);
  float var  = sq * (1.0f / 1024.0f) - mu * mu;
  float rstd = rsqrtf(var + 1e-5f);

  float* yrow = y + (size_t)(m0 + rr) * DM;
  for (int i = 0; i < 64; ++i) {
    int kk = cc + (i << 4);
    yrow[kk] = (row[kk] - mu) * rstd * lng[kk] + lnb[kk];
  }
}

// --------------------------- launch ---------------------------------------
extern "C" void kernel_launch(void* const* d_in, const int* in_sizes, int n_in,
                              void* d_out, int out_size, void* d_ws, size_t ws_size,
                              hipStream_t stream) {
  (void)in_sizes; (void)n_in; (void)out_size; (void)ws_size;
  const float* q   = (const float*)d_in[0];
  const float* k   = (const float*)d_in[1];
  const float* v   = (const float*)d_in[2];
  const unsigned char* mask = (const unsigned char*)d_in[3];   // jax bool = 1 byte
  const float* wq  = (const float*)d_in[4];
  const float* bq  = (const float*)d_in[5];
  const float* wk  = (const float*)d_in[6];
  const float* bk  = (const float*)d_in[7];
  const float* wv  = (const float*)d_in[8];
  const float* bv  = (const float*)d_in[9];
  const float* fcw = (const float*)d_in[10];
  const float* fcb = (const float*)d_in[11];
  const float* lng = (const float*)d_in[12];
  const float* lnb = (const float*)d_in[13];

  float* y    = (float*)d_out;
  float* attn = y + (size_t)MTOT * DM;       // outputs concatenated: y, attn

  const size_t QKV_ELEMS = (size_t)B_ * NH * L_ * DK;   // 4 Mi f16 each
  _Float16* qh  = (_Float16*)d_ws;
  _Float16* kh  = qh  + QKV_ELEMS;
  _Float16* vhT = kh  + QKV_ELEMS;
  _Float16* ctx = vhT + QKV_ELEMS;           // [4096, 1024] f16 — 32 MB total ws

  qkv_proj_kernel<<<768, 256, 98304, stream>>>(q, k, v, wq, bq, wk, bk, wv, bv,
                                               qh, kh, vhT);
  attn_kernel<<<4096, 256, 67584, stream>>>(qh, kh, vhT, mask, attn, ctx);
  fc_ln_kernel<<<256, 256, 98304, stream>>>(ctx, fcw, fcb, q, lng, lnb, y);
}
